// Networks_66838281060556
// MI455X (gfx1250) — matrix operand order repaired and checked by reading, not compiled, not run on hardware
//
#include <hip/hip_runtime.h>
#include <hip/hip_bf16.h>
#include <math.h>

// ---------------------------------------------------------------------------
// CDNA5 (gfx1250): all GEMMs via v_wmma_f32_16x16x32_bf16 (wave32, f32 acc).
// Operands pre-converted ONCE to padded bf16 (cols mult-of-32, zero-filled),
// so the K-loop is guard-free. A-tile and transposed-B-tile staging uses
// GLOBAL_LOAD_ASYNC_TO_LDS_B128 (ASYNCcnt) -- no VGPR round-trip, no per-chunk
// loadcnt stalls. Macro tile 256x64, 8 waves each owning a 32x64 strip
// (8 WMMAs per 6 fragment ds_load_b128). Chained intermediates emitted as
// bf16 by the GEMM epilogue. bf16 g0/g1 live in adj0/adj1 output regions,
// x0b/x1b in rec0/rec1 (overwritten only after consumption).
// ---------------------------------------------------------------------------

typedef __attribute__((ext_vector_type(16))) __bf16 v16bf;
typedef __attribute__((ext_vector_type(8)))  float  v8f;

static __device__ __forceinline__ unsigned int f2bf(float f) {
    union { float f; unsigned int u; } v; v.f = f;
    unsigned int u = v.u;
    u += 0x7FFFu + ((u >> 16) & 1u);   // round-to-nearest-even
    return u >> 16;
}
static __device__ __forceinline__ unsigned int pack2(float a, float b) {
    return f2bf(a) | (f2bf(b) << 16);
}
static __device__ __forceinline__ float fast_tanh(float x) {
    float e = __expf(2.0f * x);        // inf/0 saturate correctly to +-1
    return 1.0f - 2.0f / (e + 1.0f);
}

// async global->LDS 128-bit copy; LDS byte offset = low 32 bits of flat addr
static __device__ __forceinline__ void async_copy_b128(const void* lds_ptr,
                                                       const void* gptr) {
    unsigned lds = (unsigned)(size_t)lds_ptr;
    asm volatile("global_load_async_to_lds_b128 %0, %1, off"
                 :: "v"(lds), "v"(gptr) : "memory");
}
static __device__ __forceinline__ void wait_async0() {
    asm volatile("s_wait_asynccnt 0x0" ::: "memory");
}

union AFrag { uint4 q[2]; unsigned int u[8]; v16bf v; };
union CFrag { v8f v; float f[8]; };
union U4S8  { uint4 q; unsigned short s[8]; };

// ---------------------------------------------------------------------------
// fp32 [R x Cin] -> bf16 [Rpad x Cpad], zero-filled padding (run once/operand)
__global__ void conv_pad_kernel(const float* __restrict__ src, int R, int Cin,
                                unsigned short* __restrict__ dst, int Cpad,
                                int total8)
{
    int c = blockIdx.x * blockDim.x + threadIdx.x;
    if (c >= total8) return;
    int cpr = Cpad >> 3;
    int r   = c / cpr;
    int c0  = (c - r * cpr) << 3;
    unsigned int w[4];
#pragma unroll
    for (int j = 0; j < 4; j++) {
        int col0 = c0 + 2 * j, col1 = col0 + 1;
        float f0 = (r < R && col0 < Cin) ? src[(size_t)r * Cin + col0] : 0.0f;
        float f1 = (r < R && col1 < Cin) ? src[(size_t)r * Cin + col1] : 0.0f;
        w[j] = pack2(f0, f1);
    }
    uint4 out; out.x = w[0]; out.y = w[1]; out.z = w[2]; out.w = w[3];
    *(uint4*)&dst[(size_t)r * Cpad + c0] = out;
}

// ---------------------------------------------------------------------------
// C = act(A @ op(B) + bias); A,B bf16 (padded, zero-filled), outputs:
//   Cf (fp32, pitch ldcf) and/or Cb (bf16, pitch ldcb), pad cols zero-filled.
//   TRANSB==0: B is [Kpad x Npad] bf16   TRANSB==1: B is [* x K] bf16, use B^T
// Preconditions: M%256==0, K%32==0, lda/ldb mult of 32, grid.x*64 == Npad.
template <int ACT, int TRANSB>
__global__ __launch_bounds__(256)
void gemm_wmma_bf16(const unsigned short* __restrict__ A, int lda,
                    const unsigned short* __restrict__ B, int ldb, int Nb,
                    const float* __restrict__ bias,
                    float* __restrict__ Cf, int ldcf,
                    unsigned short* __restrict__ Cb, int ldcb,
                    int M, int K)
{
    constexpr int BM = 256, BN = 64, KT = 32;
    constexpr int PT = 40;                       // LDS pitch in ushorts (80 B)
    __shared__ __align__(16) unsigned short As[BM * PT];   // 20.0 KB
    __shared__ __align__(16) unsigned short Bs[BN * PT];   //  5.0 KB, [n][k]

    const int t     = threadIdx.x;
    const int lane  = t & 31;
    const int wave  = t >> 5;                    // 8 waves, 32-row strips
    const int row0  = blockIdx.y * BM;
    const int col0  = blockIdx.x * BN;

    const int mlane   = lane & 15;
    const int half    = lane >> 4;
    const int halfK8  = half * 8;
    const int halfK16 = half * 16;

    CFrag acc[2][4];
#pragma unroll
    for (int i = 0; i < 2; i++)
#pragma unroll
        for (int j = 0; j < 4; j++)
#pragma unroll
            for (int r = 0; r < 8; r++) acc[i][j].f[r] = 0.0f;

    for (int k0 = 0; k0 < K; k0 += KT) {
        // ---- A tile 256x32 bf16: 4x async b128 global->LDS (no VGPRs) ----
#pragma unroll
        for (int i = 0; i < 4; i++) {
            int c  = t + i * 256;                // 1024 ushort8 chunks
            int r  = c >> 2, c8 = c & 3;
            async_copy_b128(&As[r * PT + c8 * 8],
                            &A[(size_t)(row0 + r) * lda + k0 + c8 * 8]);
        }
        // ---- B tile 32x64 bf16 ----
        if (TRANSB == 0) {
            // one b128 row-chunk per thread, transpose-on-store (8x b16)
            int kk = t >> 3, n8 = t & 7;
            U4S8 v;
            v.q = *(const uint4*)&B[(size_t)(k0 + kk) * ldb + col0 + n8 * 8];
#pragma unroll
            for (int j = 0; j < 8; j++)
                Bs[(n8 * 8 + j) * PT + kk] = v.s[j];
        } else {
            // B^T: rows are C-columns, K-contiguous -> async b128
            int n = t >> 2, k8 = t & 3;
            async_copy_b128(&Bs[n * PT + k8 * 8],
                            &B[(size_t)(col0 + n) * ldb + k0 + k8 * 8]);
        }
        if (k0 + KT < K) {
            __builtin_prefetch(&A[(size_t)(row0 + t) * lda + (k0 + KT)], 0, 3);
        }
        wait_async0();                           // s_wait_asynccnt 0
        __syncthreads();

        // ---- fragments: two aligned ds_load_b128 each ----
        AFrag af[2], bfr[4];
#pragma unroll
        for (int i = 0; i < 2; i++) {
            int arow = (wave * 32 + i * 16 + mlane) * PT;
            af[i].q[0] = *(const uint4*)&As[arow + halfK8];
            af[i].q[1] = *(const uint4*)&As[arow + 16 + halfK8];
        }
#pragma unroll
        for (int j = 0; j < 4; j++) {
            int brow = (j * 16 + mlane) * PT;
            bfr[j].q[0] = *(const uint4*)&Bs[brow + halfK8];
            bfr[j].q[1] = *(const uint4*)&Bs[brow + 16 + halfK8];
        }
        // ---- 8 WMMAs: this wave's 32x64 tile ----
#pragma unroll
        for (int i = 0; i < 2; i++)
#pragma unroll
            for (int j = 0; j < 4; j++)
                acc[i][j].v = __builtin_amdgcn_wmma_f32_16x16x32_bf16(
                    false, af[i].v, false, bfr[j].v,
                    (short)0, acc[i][j].v, false, false);
        __syncthreads();
    }

    // ---- epilogue ----
#pragma unroll
    for (int i = 0; i < 2; i++) {
#pragma unroll
        for (int j = 0; j < 4; j++) {
            int grow = row0 + wave * 32 + i * 16 + (half << 3);
            int gcol = col0 + j * 16 + mlane;
            float bval = bias ? bias[gcol < Nb ? gcol : 0] : 0.0f;
#pragma unroll
            for (int r = 0; r < 8; r++) {
                float v = acc[i][j].f[r] + bval;
                if (ACT == 1) v = fast_tanh(v);
                else if (ACT == 2) v = 1.0f / (1.0f + __expf(-v));
                if (gcol >= Nb) v = 0.0f;        // zero-fill pad columns
                if (Cf) Cf[(size_t)(grow + r) * ldcf + gcol] = v;
                if (Cb) Cb[(size_t)(grow + r) * ldcb + gcol] =
                            (unsigned short)f2bf(v);
            }
        }
    }
}

// ---------------------------------------------------------------------------
// z = (we0*h0+we1*h1)/(we0+we1); writes fp32 zbuf (pitch P), bf16 zb, tight z
__global__ void fuse_z_kernel(const float* __restrict__ h0,
                              const float* __restrict__ h1,
                              const float* __restrict__ we,
                              float* __restrict__ z_ws,
                              unsigned short* __restrict__ z_bf,
                              float* __restrict__ z_out,
                              int Nn, int S, int P)
{
    int idx = blockIdx.x * blockDim.x + threadIdx.x;
    if (idx >= Nn * P) return;
    int row = idx / P, col = idx - row * P;
    float v = 0.0f;
    if (col < S) {
        float w0 = we[row * 2 + 0];
        float w1 = we[row * 2 + 1];
        v = (w0 * h0[idx] + w1 * h1[idx]) / (w0 + w1);
        z_out[(size_t)row * S + col] = v;
    }
    z_ws[idx] = v;
    z_bf[idx] = (unsigned short)f2bf(v);
}

__global__ void zero_kernel(float* __restrict__ p, int n)
{
    int i = blockIdx.x * blockDim.x + threadIdx.x;
    if (i < n) p[i] = 0.0f;
}

__global__ void q_kernel(const float* __restrict__ z, const float* __restrict__ centers,
                         float* __restrict__ q, float* __restrict__ colsum,
                         int Nn, int S, int P, int Kc)
{
    int row = blockIdx.x * blockDim.x + threadIdx.x;
    if (row >= Nn) return;
    float qs[8];
    float ssum = 0.0f;
    for (int k = 0; k < Kc; k++) {
        float d2 = 0.0f;
        const float* zr = z + (size_t)row * P;
        const float* cr = centers + (size_t)k * S;
        for (int s = 0; s < S; s++) {
            float d = zr[s] - cr[s];
            d2 += d * d;
        }
        float qq = 1.0f / (1.0f + d2);
        qs[k] = qq;
        ssum += qq;
    }
    for (int k = 0; k < Kc; k++) {
        float qn = qs[k] / ssum;
        q[row * Kc + k] = qn;
        atomicAdd(&colsum[k], qn);
    }
}

__global__ void p_kernel(const float* __restrict__ q, const float* __restrict__ colsum,
                         float* __restrict__ p, int Nn, int Kc)
{
    int row = blockIdx.x * blockDim.x + threadIdx.x;
    if (row >= Nn) return;
    float w[8];
    float s = 0.0f;
    for (int k = 0; k < Kc; k++) {
        float v = q[row * Kc + k];
        w[k] = v * v / colsum[k];
        s += w[k];
    }
    for (int k = 0; k < Kc; k++) p[row * Kc + k] = w[k] / s;
}

// ---------------------------------------------------------------------------

static void conv(const float* src, int R, int Cin, unsigned short* dst,
                 int Rpad, int Cpad, hipStream_t s)
{
    int total8 = Rpad * Cpad / 8;
    conv_pad_kernel<<<(total8 + 255) / 256, 256, 0, s>>>(src, R, Cin, dst, Cpad, total8);
}

static void gemm(const unsigned short* A, int lda,
                 const unsigned short* B, int ldb, int Nb,
                 const float* bias, float* Cf, int ldcf,
                 unsigned short* Cb, int ldcb,
                 int M, int K, int Npad, int act, int transB, hipStream_t s)
{
    dim3 grid(Npad / 64, M / 256);
    dim3 block(256);
    if (transB == 0) {
        if (act == 0)
            gemm_wmma_bf16<0, 0><<<grid, block, 0, s>>>(A, lda, B, ldb, Nb, bias,
                                                        Cf, ldcf, Cb, ldcb, M, K);
        else
            gemm_wmma_bf16<1, 0><<<grid, block, 0, s>>>(A, lda, B, ldb, Nb, bias,
                                                        Cf, ldcf, Cb, ldcb, M, K);
    } else {
        gemm_wmma_bf16<2, 1><<<grid, block, 0, s>>>(A, lda, B, ldb, Nb, bias,
                                                    Cf, ldcf, Cb, ldcb, M, K);
    }
}

extern "C" void kernel_launch(void* const* d_in, const int* in_sizes, int n_in,
                              void* d_out, int out_size, void* d_ws, size_t ws_size,
                              hipStream_t stream)
{
    constexpr int N = 4096, D0 = 1024, D1 = 1280, H1 = 1024, S = 819, K = 5;
    constexpr int SP = 832;   // 819 -> multiple of 32

    const float* x0   = (const float*)d_in[0];
    const float* x1   = (const float*)d_in[1];
    const float* we   = (const float*)d_in[2];
    const float* g0   = (const float*)d_in[3];
    const float* g1   = (const float*)d_in[4];
    const float* We1_0 = (const float*)d_in[5];  const float* be1_0 = (const float*)d_in[6];
    const float* We2_0 = (const float*)d_in[7];  const float* be2_0 = (const float*)d_in[8];
    const float* We1_1 = (const float*)d_in[9];  const float* be1_1 = (const float*)d_in[10];
    const float* We2_1 = (const float*)d_in[11]; const float* be2_1 = (const float*)d_in[12];
    const float* Wd1_0 = (const float*)d_in[13]; const float* bd1_0 = (const float*)d_in[14];
    const float* Wd2_0 = (const float*)d_in[15]; const float* bd2_0 = (const float*)d_in[16];
    const float* Wd1_1 = (const float*)d_in[17]; const float* bd1_1 = (const float*)d_in[18];
    const float* Wd2_1 = (const float*)d_in[19]; const float* bd2_1 = (const float*)d_in[20];
    const float* Ww_0  = (const float*)d_in[21];
    const float* Ww_1  = (const float*)d_in[22];
    const float* Wf_0  = (const float*)d_in[23];
    const float* Wf_1  = (const float*)d_in[24];
    const float* clus  = (const float*)d_in[25];

    // ---- outputs (fp32), concatenated in reference return order ----
    float* out  = (float*)d_out;
    float* rec0 = out;
    float* rec1 = rec0 + (size_t)N * D0;
    float* adj0 = rec1 + (size_t)N * D1;
    float* adj1 = adj0 + (size_t)N * N;
    float* zout = adj1 + (size_t)N * N;
    float* pout = zout + (size_t)N * S;
    float* qout = pout + (size_t)N * K;
    float* cout = qout + (size_t)N * K;

    // bf16 tenants inside output regions (consumed before region is written):
    unsigned short* g0b = (unsigned short*)adj0;   // 33.5MB in 67MB region
    unsigned short* g1b = (unsigned short*)adj1;
    unsigned short* x0b = (unsigned short*)rec0;   // 8.4MB in 16.8MB region
    unsigned short* x1b = (unsigned short*)rec1;

    // ---- workspace ----
    float* ws = (float*)d_ws;
    const size_t SZ_NS = (size_t)N * SP;
    float* h2_0   = ws;
    float* h2_1   = h2_0 + SZ_NS;
    float* zbuf   = h2_1 + SZ_NS;
    float* colsum = zbuf + SZ_NS;                  // K floats (+ pad to 16)
    unsigned short* bws = (unsigned short*)(colsum + 16);
    unsigned short* tAb = bws;                 bws += (size_t)N * 1280;
    unsigned short* tBb = bws;                 bws += (size_t)N * 1280;
    unsigned short* zb  = bws;                 bws += (size_t)N * SP;
    unsigned short* We1_0b = bws;              bws += (size_t)1024 * SP;
    unsigned short* We2_0b = bws;              bws += (size_t)SP * SP;
    unsigned short* We1_1b = bws;              bws += (size_t)1280 * 1024;
    unsigned short* We2_1b = bws;              bws += (size_t)1024 * SP;
    unsigned short* Wd1_0b = bws;              bws += (size_t)SP * SP;
    unsigned short* Wd2_0b = bws;              bws += (size_t)SP * 1024;
    unsigned short* Wd1_1b = bws;              bws += (size_t)SP * 1024;
    unsigned short* Wd2_1b = bws;              bws += (size_t)1024 * 1280;
    unsigned short* Ww_0b  = bws;              bws += (size_t)SP * SP;
    unsigned short* Ww_1b  = bws;              bws += (size_t)SP * SP;
    unsigned short* Wf_0b  = bws;              bws += (size_t)SP * SP;
    unsigned short* Wf_1b  = bws;              bws += (size_t)SP * SP;

    // ---------------- one-time bf16 conversions ----------------
    conv(g0, N, N,        g0b,    N,    N,    stream);
    conv(g1, N, N,        g1b,    N,    N,    stream);
    conv(x0, N, D0,       x0b,    N,    D0,   stream);
    conv(x1, N, D1,       x1b,    N,    D1,   stream);
    conv(We1_0, D0, S,    We1_0b, D0,   SP,   stream);
    conv(We2_0, S,  S,    We2_0b, SP,   SP,   stream);
    conv(We1_1, D1, H1,   We1_1b, D1,   H1,   stream);
    conv(We2_1, H1, S,    We2_1b, H1,   SP,   stream);
    conv(Wd1_0, S,  S,    Wd1_0b, SP,   SP,   stream);
    conv(Wd2_0, S,  D0,   Wd2_0b, SP,   D0,   stream);
    conv(Wd1_1, S,  H1,   Wd1_1b, SP,   H1,   stream);
    conv(Wd2_1, H1, D1,   Wd2_1b, H1,   D1,   stream);
    conv(Ww_0,  S,  S,    Ww_0b,  SP,   SP,   stream);
    conv(Ww_1,  S,  S,    Ww_1b,  SP,   SP,   stream);
    conv(Wf_0,  S,  S,    Wf_0b,  SP,   SP,   stream);
    conv(Wf_1,  S,  S,    Wf_1b,  SP,   SP,   stream);

    // ---------------- encoders ----------------
    gemm(g0b, N,   x0b, D0, D0,     nullptr, nullptr, 0, tAb, D0,  N, N,   D0, 0, 0, stream);
    gemm(tAb, D0,  We1_0b, SP, S,   be1_0,   nullptr, 0, tBb, SP,  N, D0,  SP, 1, 0, stream);
    gemm(g0b, N,   tBb, SP, SP,     nullptr, nullptr, 0, tAb, SP,  N, N,   SP, 0, 0, stream);
    gemm(tAb, SP,  We2_0b, SP, S,   be2_0,   h2_0, SP, nullptr, 0, N, SP,  SP, 1, 0, stream);
    gemm(g1b, N,   x1b, D1, D1,     nullptr, nullptr, 0, tAb, D1,  N, N,   D1, 0, 0, stream);
    gemm(tAb, D1,  We1_1b, H1, H1,  be1_1,   nullptr, 0, tBb, H1,  N, D1,  H1, 1, 0, stream);
    gemm(g1b, N,   tBb, H1, H1,     nullptr, nullptr, 0, tAb, H1,  N, N,   H1, 0, 0, stream);
    gemm(tAb, H1,  We2_1b, SP, S,   be2_1,   h2_1, SP, nullptr, 0, N, H1,  SP, 1, 0, stream);

    // ---------------- z fusion ----------------
    {
        int total = N * SP;
        fuse_z_kernel<<<(total + 255) / 256, 256, 0, stream>>>(h2_0, h2_1, we,
                                                               zbuf, zb, zout, N, S, SP);
    }

    // ---------------- feature decoders (need g0b/g1b -> run before adj) ----
    gemm(zb, SP,   Wf_0b, SP, S,    nullptr, nullptr, 0, tAb, SP,  N, SP,  SP, 1, 0, stream);
    gemm(g0b, N,   tAb, SP, SP,     nullptr, nullptr, 0, tBb, SP,  N, N,   SP, 0, 0, stream);
    gemm(tBb, SP,  Wd1_0b, SP, S,   bd1_0,   nullptr, 0, tAb, SP,  N, SP,  SP, 1, 0, stream);
    gemm(g0b, N,   tAb, SP, SP,     nullptr, nullptr, 0, tBb, SP,  N, N,   SP, 0, 0, stream);
    gemm(tBb, SP,  Wd2_0b, D0, D0,  bd2_0,   rec0, D0, nullptr, 0, N, SP,  D0, 1, 0, stream);

    gemm(zb, SP,   Wf_1b, SP, S,    nullptr, nullptr, 0, tAb, SP,  N, SP,  SP, 1, 0, stream);
    gemm(g1b, N,   tAb, SP, SP,     nullptr, nullptr, 0, tBb, SP,  N, N,   SP, 0, 0, stream);
    gemm(tBb, SP,  Wd1_1b, H1, H1,  bd1_1,   nullptr, 0, tAb, H1,  N, SP,  H1, 1, 0, stream);
    gemm(g1b, N,   tAb, H1, H1,     nullptr, nullptr, 0, tBb, H1,  N, N,   H1, 0, 0, stream);
    gemm(tBb, H1,  Wd2_1b, D1, D1,  bd2_1,   rec1, D1, nullptr, 0, N, H1,  D1, 1, 0, stream);

    // ---------------- adjacency (last: overwrites g0b/g1b regions) ----------
    gemm(zb, SP,   Ww_0b, SP, S,    nullptr, nullptr, 0, tAb, SP,  N, SP,  SP, 0, 0, stream);
    gemm(tAb, SP,  zb, SP, N,       nullptr, adj0, N,  nullptr, 0, N, SP,  N,  2, 1, stream);
    gemm(zb, SP,   Ww_1b, SP, S,    nullptr, nullptr, 0, tAb, SP,  N, SP,  SP, 0, 0, stream);
    gemm(tAb, SP,  zb, SP, N,       nullptr, adj1, N,  nullptr, 0, N, SP,  N,  2, 1, stream);

    // ---------------- Student-t soft assignment ----------------
    zero_kernel<<<1, 32, 0, stream>>>(colsum, K);
    q_kernel<<<(N + 127) / 128, 128, 0, stream>>>(zbuf, clus, qout, colsum, N, S, SP, K);
    p_kernel<<<(N + 127) / 128, 128, 0, stream>>>(qout, colsum, pout, N, K);

    // ---------------- cluster_layer passthrough ----------------
    hipMemcpyAsync(cout, clus, (size_t)K * S * sizeof(float),
                   hipMemcpyDeviceToDevice, stream);
}
